// Encoder_2834678415877
// MI455X (gfx1250) — matrix operand order repaired and checked by reading, not compiled, not run on hardware
//
#include <hip/hip_runtime.h>
#include <math.h>

// CDNA5 (gfx1250) wave32 WMMA types
typedef __attribute__((ext_vector_type(16))) _Float16 v16h;
typedef __attribute__((ext_vector_type(8)))  float    v8f;
typedef __attribute__((ext_vector_type(4)))  float    f32x4;

// ---------------------------------------------------------------------------
// Pack W (K x N fp32, row-major) into f16 WMMA B-fragment order:
//   frag f = (tN*KS + ks), lane l, slot j  ->  W[ks*32 + (l>>4)*8 + (j&7) + (j>>3)*16, tN*16 + (l&15)]
// stored at Wp[((tN*KS + ks)*32 + l)*16 + j].  One thread per packed element.
// Total elements = N*K (always a multiple of 256 here).
// ---------------------------------------------------------------------------
__global__ void __launch_bounds__(256)
pack_w_f16(const float* __restrict__ W, _Float16* __restrict__ Wp, int K, int N)
{
    const int t    = blockIdx.x * 256 + threadIdx.x;
    const int j    = t & 15;
    const int lane = (t >> 4) & 31;
    const int rest = t >> 9;
    const int KS   = K >> 5;
    const int ks   = rest % KS;
    const int tN   = rest / KS;
    const int n    = tN * 16 + (lane & 15);
    const int k    = ks * 32 + (lane >> 4) * 8 + (j & 7) + (j >> 3) * 16;
    Wp[t] = (_Float16)W[(size_t)k * N + n];
}

// ---------------------------------------------------------------------------
// Y[M,N] = X[M,K] @ W[K,N] + bias[N]
// fp32 activations converted to f16 on the fly; W pre-packed f16 fragments.
// One wave per 32x32 output tile: 2 A-frags x 2 B-frags -> 4 WMMAs / k-step.
//
// A fragment (16x32 f16, ISA 7.12.2): lane l -> row m = l&15; koff=(l<16)?0:8;
//   a[0..7] = K = koff+0..7, a[8..15] = K = 16+koff+0..7  (two aligned f32x4
//   pairs per run).
// C/D (16x16 f32): VGPR r, lane l -> (row r + ((l<16)?0:8), col l&15).
// ---------------------------------------------------------------------------
__global__ void __launch_bounds__(32)
wmma_gemm_bias(const float* __restrict__ X, const _Float16* __restrict__ Wp,
               const float* __restrict__ bias, float* __restrict__ Y,
               int M, int K, int N)
{
    const int lane = threadIdx.x & 31;
    const int l16  = lane & 15;
    const int half = lane >> 4;
    const int bm   = blockIdx.x;          // 32-row block
    const int bn   = blockIdx.y;          // 32-col block
    const int KS   = K >> 5;

    const int n0 = bn * 32 + l16;
    const int n1 = n0 + 16;

    v8f acc00, acc01, acc10, acc11;
    const float c0 = bias ? bias[n0] : 0.0f;
    const float c1 = bias ? bias[n1] : 0.0f;
#pragma unroll
    for (int r = 0; r < 8; ++r) {
        acc00[r] = c0; acc01[r] = c1;
        acc10[r] = c0; acc11[r] = c1;
    }

    const int m0 = bm * 32 + l16;
    const f32x4* __restrict__ xr0 = (const f32x4*)(X + (size_t)m0 * K);
    const f32x4* __restrict__ xr1 = (const f32x4*)(X + (size_t)(m0 + 16) * K);
    const v16h*  __restrict__ Wv  = (const v16h*)Wp;
    const size_t wb0 = ((size_t)(bn * 2) * KS) * 32 + lane;      // B frag 0 base
    const size_t wb1 = wb0 + (size_t)KS * 32;                    // B frag 1 base

    for (int ks = 0; ks < KS; ++ks) {
        const int q = ks * 8 + half * 2;          // f32x4 index of a[0..3]
        __builtin_prefetch(xr0 + q + 8, 0, 1);    // speculative (global_prefetch_b8)

        const f32x4 p0 = xr0[q], p1 = xr0[q + 1], p2 = xr0[q + 4], p3 = xr0[q + 5];
        const f32x4 r0 = xr1[q], r1 = xr1[q + 1], r2 = xr1[q + 4], r3 = xr1[q + 5];

        v16h a0, a1;
#pragma unroll
        for (int i = 0; i < 4; ++i) {
            a0[i]      = (_Float16)p0[i];
            a0[i + 4]  = (_Float16)p1[i];
            a0[i + 8]  = (_Float16)p2[i];
            a0[i + 12] = (_Float16)p3[i];
            a1[i]      = (_Float16)r0[i];
            a1[i + 4]  = (_Float16)r1[i];
            a1[i + 8]  = (_Float16)r2[i];
            a1[i + 12] = (_Float16)r3[i];
        }

        const v16h b0 = Wv[wb0 + (size_t)ks * 32];
        const v16h b1 = Wv[wb1 + (size_t)ks * 32];

        acc00 = __builtin_amdgcn_wmma_f32_16x16x32_f16(false, a0, false, b0,
                                                       (short)0, acc00, false, false);
        acc01 = __builtin_amdgcn_wmma_f32_16x16x32_f16(false, a0, false, b1,
                                                       (short)0, acc01, false, false);
        acc10 = __builtin_amdgcn_wmma_f32_16x16x32_f16(false, a1, false, b0,
                                                       (short)0, acc10, false, false);
        acc11 = __builtin_amdgcn_wmma_f32_16x16x32_f16(false, a1, false, b1,
                                                       (short)0, acc11, false, false);
    }

#pragma unroll
    for (int r = 0; r < 8; ++r) {
        const size_t row0 = (size_t)(bm * 32 + r + half * 8);
        Y[row0 * N + n0]        = acc00[r];
        Y[row0 * N + n1]        = acc01[r];
        Y[(row0 + 16) * N + n0] = acc10[r];
        Y[(row0 + 16) * N + n1] = acc11[r];
    }
}

// ---------------------------------------------------------------------------
// Neighborhood attention: one wave32 per token (b,n).
//   scores_j = (q . k[nbr_j] + rb[n%8, j]) / sqrt(C);  softmax over j=0..7;
//   out = sum_j p_j * v[nbr_j].
// O may alias Q (each wave fully consumes its own q row before writing out).
// ---------------------------------------------------------------------------
__global__ void __launch_bounds__(256)
na_attn(const float* __restrict__ Q, const float* __restrict__ Kb,
        const float* __restrict__ V, const int* __restrict__ nbrs,
        const float* __restrict__ rb, float* __restrict__ O,
        int Ntok, int Ntotal, int C)
{
    const int wave = (int)((blockIdx.x * blockDim.x + threadIdx.x) >> 5);
    const int lane = threadIdx.x & 31;
    if (wave >= Ntotal) return;

    const int n    = wave % Ntok;
    const int base = wave - n;

    int idx[8];
#pragma unroll
    for (int j = 0; j < 8; ++j) idx[j] = base + nbrs[n * 8 + j];

    const float* __restrict__ qrow = Q + (size_t)wave * C;
    const float scale = rsqrtf((float)C);

    float s[8];
#pragma unroll
    for (int j = 0; j < 8; ++j) {
        const float* __restrict__ krow = Kb + (size_t)idx[j] * C;
        float p = 0.0f;
        for (int c = lane; c < C; c += 32) p += qrow[c] * krow[c];
        for (int o = 16; o > 0; o >>= 1) p += __shfl_xor(p, o, 32);
        s[j] = (p + rb[(n & 7) * 8 + j]) * scale;
    }

    float mx = s[0];
#pragma unroll
    for (int j = 1; j < 8; ++j) mx = fmaxf(mx, s[j]);
    float sum = 0.0f;
#pragma unroll
    for (int j = 0; j < 8; ++j) { s[j] = __expf(s[j] - mx); sum += s[j]; }
    const float inv = 1.0f / sum;

    float* __restrict__ orow = O + (size_t)wave * C;
    for (int c = lane; c < C; c += 32) {
        float acc = 0.0f;
#pragma unroll
        for (int j = 0; j < 8; ++j) acc += s[j] * V[(size_t)idx[j] * C + c];
        orow[c] = acc * inv;
    }
}

// ---------------------------------------------------------------------------
// 2x2 patch merge + LayerNorm: one wave32 per merged token.
// ---------------------------------------------------------------------------
__global__ void __launch_bounds__(256)
merge_ln(const float* __restrict__ T, const float* __restrict__ g,
         const float* __restrict__ be, float* __restrict__ Y,
         int B, int H, int Wd, int C)
{
    const int wave = (int)((blockIdx.x * blockDim.x + threadIdx.x) >> 5);
    const int lane = threadIdx.x & 31;
    const int H2 = H >> 1, W2 = Wd >> 1;
    const int C4 = 4 * C;
    const int tot = B * H2 * W2;
    if (wave >= tot) return;

    const int b  = wave / (H2 * W2);
    const int r  = wave % (H2 * W2);
    const int h2 = r / W2, w2 = r % W2;

    auto getv = [&](int cc) -> float {
        const int sub = cc / C, c = cc % C;
        const int i2 = sub >> 1, j2 = sub & 1;
        return T[((size_t)(b * H + 2 * h2 + i2) * Wd + (2 * w2 + j2)) * C + c];
    };

    float sum = 0.0f, ss = 0.0f;
    for (int cc = lane; cc < C4; cc += 32) {
        const float v = getv(cc);
        sum += v; ss += v * v;
    }
    for (int o = 16; o > 0; o >>= 1) {
        sum += __shfl_xor(sum, o, 32);
        ss  += __shfl_xor(ss,  o, 32);
    }
    const float mu  = sum / (float)C4;
    const float var = ss / (float)C4 - mu * mu;
    const float inv = rsqrtf(var + 1e-5f);

    float* __restrict__ yrow = Y + (size_t)wave * C4;
    for (int cc = lane; cc < C4; cc += 32) {
        yrow[cc] = (getv(cc) - mu) * inv * g[cc] + be[cc];
    }
}

// ---------------------------------------------------------------------------
static inline void run_gemm(const float* X, const float* W, const float* bias,
                            float* Y, _Float16* Wpack,
                            int M, int K, int N, hipStream_t s)
{
    pack_w_f16<<<dim3((K * N) / 256), 256, 0, s>>>(W, Wpack, K, N);
    dim3 grid(M / 32, N / 32);
    wmma_gemm_bias<<<grid, 32, 0, s>>>(X, Wpack, bias, Y, M, K, N);
}

extern "C" void kernel_launch(void* const* d_in, const int* in_sizes, int n_in,
                              void* d_out, int out_size, void* d_ws, size_t ws_size,
                              hipStream_t stream)
{
    (void)in_sizes; (void)n_in; (void)out_size; (void)ws_size;

    const float* x0 = (const float*)d_in[0];
    const int* nb[3] = { (const int*)d_in[1], (const int*)d_in[2], (const int*)d_in[3] };

    // params flattened in dict insertion order:
    // per stage s: wq bq wk bk wv bv wo bo rb g be wr  -> 12 tensors
    auto pw = [&](int s, int j) { return (const float*)d_in[4 + s * 12 + j]; };

    float* out = (float*)d_out;
    const int B = 4;
    const size_t S = (size_t)B * 128 * 128 * 96;   // 6,291,456 floats per region

    float* R0 = (float*)d_ws;
    float* R1 = R0 + S;
    float* R2 = R1 + S;
    float* R3 = R2 + S;
    _Float16* Wpack = (_Float16*)(R3 + S);         // packed weights (<= 1.18M f16)

    float* xfinal  = out;                                      // (4,16,16,768)
    float* skip[3] = { out + 786432,                           // (4,16384, 96)
                       out + 786432 + 6291456,                 // (4, 4096,192)
                       out + 786432 + 6291456 + 3145728 };     // (4, 1024,384)

    const float* xcur = x0;
    for (int s = 0; s < 3; ++s) {
        const int Hs   = 128 >> s;
        const int Ws   = 128 >> s;
        const int Cs   = 96 << s;
        const int Ntok = Hs * Ws;
        const int M    = B * Ntok;

        float* q = R0;
        float* k = R1;
        float* v = (s == 1) ? R3 : R2;      // never overlaps current xcur

        // Q/K/V projections  (M x Cs x Cs each, WMMA)
        run_gemm(xcur, pw(s, 0), pw(s, 1), q, Wpack, M, Cs, Cs, stream);
        run_gemm(xcur, pw(s, 2), pw(s, 3), k, Wpack, M, Cs, Cs, stream);
        run_gemm(xcur, pw(s, 4), pw(s, 5), v, Wpack, M, Cs, Cs, stream);

        // neighborhood attention, in-place over q
        na_attn<<<dim3(M / 8), 256, 0, stream>>>(q, k, v, nb[s], pw(s, 8),
                                                 q, Ntok, M, Cs);

        // output projection -> skip output (this stage's returned tensor)
        run_gemm(q, pw(s, 6), pw(s, 7), skip[s], Wpack, M, Cs, Cs, stream);

        // 2x2 patch merge + LayerNorm -> merged (M/4, 4*Cs), over dead k
        float* merged = R1;
        const int mtok = B * (Hs / 2) * (Ws / 2);
        merge_ln<<<dim3(mtok / 8), 256, 0, stream>>>(skip[s], pw(s, 9), pw(s, 10),
                                                     merged, B, Hs, Ws, Cs);

        // channel reduction (4C -> 2C), no bias; final stage writes x_final
        float* nx = (s == 0) ? R2 : (s == 1) ? R3 : xfinal;
        run_gemm(merged, pw(s, 11), nullptr, nx, Wpack, M / 4, 4 * Cs, 2 * Cs, stream);

        xcur = nx;
    }
}
